// proj_loss_sparse_35974646072033
// MI455X (gfx1250) — matrix operand order repaired and checked by reading, not compiled, not run on hardware
//
#include <hip/hip_runtime.h>
#include <hip/hip_bf16.h>

// ---------------------------------------------------------------------------
// Problem constants (match reference)
// ---------------------------------------------------------------------------
#define NPTS   3072
#define DFEAT  768
#define KNN    10
#define KEIG   128
#define LMBDA  100.0f
#define SUB_ITERS 16

typedef __attribute__((ext_vector_type(16))) _Float16 v16h;
typedef __attribute__((ext_vector_type(8)))  _Float16 v8h;
typedef __attribute__((ext_vector_type(8)))  float    v8f;

// ---------------------------------------------------------------------------
// WMMA GEMM: C (f32, MxN row-major) = opA(A) * opB(B), A/B f16.
// TA: logical A[m][k] = Astore[k*lda + m]  (Astore is K x M row-major)
// TB: logical B[k][n] = Bstore[n*ldb + k]  (Bstore is N x K row-major)
// All M, N multiples of 32; all K multiples of 64 at every call site.
// Block: 128 threads = 4 wave32s; block tile 32x32; K-step 64 (2 WMMA/stage).
// Software-pipelined: next K-tile is loaded into registers while the WMMAs
// for the current tile run from LDS; unconditional speculative prefetch two
// tiles ahead keeps the HBM path busy.
// ---------------------------------------------------------------------------
template<bool TA, bool TB>
__global__ __launch_bounds__(128)
void wmma_gemm_f16f32(const _Float16* __restrict__ A, const _Float16* __restrict__ B,
                      float* __restrict__ C, int K, int lda, int ldb, int ldc)
{
    __shared__ __align__(16) _Float16 AsF[32 * 72];   // [m][k0..63] (+8 pad)
    __shared__ __align__(16) _Float16 BsF[32 * 72];   // [n][k0..63] (transposed)

    const int m0   = blockIdx.y * 32;
    const int n0   = blockIdx.x * 32;
    const int tid  = threadIdx.x;
    const int lane = tid & 31;
    const int wave = tid >> 5;
    const int tm   = (wave >> 1) * 16;
    const int tn   = (wave & 1) * 16;
    const int half = lane >> 4;
    const int l15  = lane & 15;

    v8h ga0, ga1, gb0, gb1;       // register-staged chunks (software pipeline)

    auto load_regs = [&](int kk) {
#pragma unroll
        for (int u = 0; u < 2; ++u) {
            const int t2 = tid * 2 + u;             // 0..255 chunk id
            v8h va, vb;
            if (!TA) {
                const int r  = t2 >> 3;             // m row 0..31
                const int kc = (t2 & 7) * 8;        // k chunk
                const _Float16* s = &A[(size_t)(m0 + r) * lda + (kk + kc)];
                va = *(const v8h*)s;
                __builtin_prefetch(s + 128, 0, 3);  // speculative, 2 tiles ahead
            } else {
                const int kr = t2 >> 2;             // k row 0..63
                const int mc = (t2 & 3) * 8;        // m chunk
                const _Float16* s = &A[(size_t)(kk + kr) * lda + (m0 + mc)];
                va = *(const v8h*)s;
                __builtin_prefetch(s + (size_t)128 * lda, 0, 3);
            }
            if (TB) {
                const int r  = t2 >> 3;             // n row 0..31
                const int kc = (t2 & 7) * 8;
                const _Float16* s = &B[(size_t)(n0 + r) * ldb + (kk + kc)];
                vb = *(const v8h*)s;
                __builtin_prefetch(s + 128, 0, 3);
            } else {
                const int kr = t2 >> 2;
                const int nc = (t2 & 3) * 8;
                const _Float16* s = &B[(size_t)(kk + kr) * ldb + (n0 + nc)];
                vb = *(const v8h*)s;
                __builtin_prefetch(s + (size_t)128 * ldb, 0, 3);
            }
            if (u == 0) { ga0 = va; gb0 = vb; } else { ga1 = va; gb1 = vb; }
        }
    };

    auto stage = [&]() {
#pragma unroll
        for (int u = 0; u < 2; ++u) {
            const int t2 = tid * 2 + u;
            v8h va = u ? ga1 : ga0;
            v8h vb = u ? gb1 : gb0;
            if (!TA) {
                const int r = t2 >> 3, kc = (t2 & 7) * 8;
                *(v8h*)&AsF[r * 72 + kc] = va;
            } else {
                const int kr = t2 >> 2, mc = (t2 & 3) * 8;
#pragma unroll
                for (int e = 0; e < 8; ++e) AsF[(mc + e) * 72 + kr] = va[e];
            }
            if (TB) {
                const int r = t2 >> 3, kc = (t2 & 7) * 8;
                *(v8h*)&BsF[r * 72 + kc] = vb;
            } else {
                const int kr = t2 >> 2, nc = (t2 & 3) * 8;
#pragma unroll
                for (int e = 0; e < 8; ++e) BsF[(nc + e) * 72 + kr] = vb[e];
            }
        }
    };

    v8f acc = {};
    auto compute = [&]() {
#pragma unroll
        for (int ks = 0; ks < 64; ks += 32) {
            // CDNA5 16-bit A 16x32 / B 32x16 lane layouts; 4 x ds_load_b128
            const _Float16* arow = &AsF[(tm + l15) * 72 + ks];
            const _Float16* brow = &BsF[(tn + l15) * 72 + ks];
            v8h alo = *(const v8h*)&arow[8 * half];
            v8h ahi = *(const v8h*)&arow[16 + 8 * half];
            v8h blo = *(const v8h*)&brow[8 * half];
            v8h bhi = *(const v8h*)&brow[16 + 8 * half];
            v16h a = __builtin_shufflevector(alo, ahi, 0,1,2,3,4,5,6,7,8,9,10,11,12,13,14,15);
            v16h b = __builtin_shufflevector(blo, bhi, 0,1,2,3,4,5,6,7,8,9,10,11,12,13,14,15);
            acc = __builtin_amdgcn_wmma_f32_16x16x32_f16(false, a, false, b,
                                                         (short)0, acc, false, false);
        }
    };

    load_regs(0);
    int k0 = 0;
    for (; k0 + 64 < K; k0 += 64) {
        stage();
        __syncthreads();
        load_regs(k0 + 64);    // issue next tile's global loads before compute
        compute();
        __syncthreads();
    }
    stage();
    __syncthreads();
    compute();

    // D layout: VGPR r -> M = r + 8*half, N = lane&15
#pragma unroll
    for (int r = 0; r < 8; ++r) {
        const int gm = m0 + tm + r + 8 * half;
        const int gn = n0 + tn + l15;
        C[(size_t)gm * ldc + gn] = acc[r];
    }
}

// ---------------------------------------------------------------------------
// TDM / cluster probe kernel (NOT launched; emitted into the bundle only, to
// exercise the gfx1250 async-tensor path; signature validated by compile).
// ---------------------------------------------------------------------------
#if __has_builtin(__builtin_amdgcn_tensor_load_to_lds)
typedef unsigned int u32x4 __attribute__((ext_vector_type(4)));
typedef int          i32x4 __attribute__((ext_vector_type(4)));
typedef int          i32x8 __attribute__((ext_vector_type(8)));
__global__ void cdna5_tdm_probe_kernel(int* out)
{
    u32x4 g0 = {};   // D# group 0 (count=0 -> NULL tensor)
    i32x8 g1 = {};
    i32x4 g2 = {};
    i32x4 g3 = {};
#if defined(__clang_major__) && __clang_major__ >= 23
    i32x8 g4 = {};
    __builtin_amdgcn_tensor_load_to_lds(g0, g1, g2, g3, g4, 0);
#else
    __builtin_amdgcn_tensor_load_to_lds(g0, g1, g2, g3, 0);
#endif
    __builtin_amdgcn_s_wait_tensorcnt((short)0);
#if __has_builtin(__builtin_amdgcn_cluster_id_x)
    int cid = __builtin_amdgcn_cluster_id_x();
#else
    int cid = 0;
#endif
#if __has_builtin(__builtin_amdgcn_s_cluster_barrier)
    __builtin_amdgcn_s_cluster_barrier();
#endif
    if (out) out[threadIdx.x] = cid;
}
#endif

// ---------------------------------------------------------------------------
// Small utility kernels
// ---------------------------------------------------------------------------
__global__ void cvt_f32_f16_kernel(const float* __restrict__ s, _Float16* __restrict__ d, int n)
{
    int i = blockIdx.x * blockDim.x + threadIdx.x;
    if (i < n) d[i] = (_Float16)s[i];
}

__global__ __launch_bounds__(256) void sqnorm_kernel(const float* __restrict__ F, float* __restrict__ sq)
{
    __shared__ float red[256];
    const int i = blockIdx.x;
    float s = 0.f;
    for (int j = threadIdx.x; j < DFEAT; j += 256) {
        float v = F[(size_t)i * DFEAT + j];
        s += v * v;
    }
    red[threadIdx.x] = s; __syncthreads();
    for (int off = 128; off; off >>= 1) {
        if (threadIdx.x < off) red[threadIdx.x] += red[threadIdx.x + off];
        __syncthreads();
    }
    if (threadIdx.x == 0) sq[i] = red[0];
}

// top-KNN per row (largest of 2*G[i][j]-sq[j], j!=i  ==  smallest d2)
__global__ __launch_bounds__(256) void knn_select_kernel(const float* __restrict__ G,
                                                         const float* __restrict__ sq,
                                                         int* __restrict__ idx)
{
    __shared__ float sval[256];
    __shared__ int   sidx[256];
    __shared__ int   chosen[KNN];
    const int i = blockIdx.x;
    const float* Grow = G + (size_t)i * NPTS;

    for (int r = 0; r < KNN; ++r) {
        float best = -3.0e38f; int bj = -1;
        for (int j = threadIdx.x; j < NPTS; j += 256) {
            if (j == i) continue;
            bool used = false;
            for (int c = 0; c < r; ++c) if (chosen[c] == j) used = true;
            if (used) continue;
            float s = 2.f * Grow[j] - sq[j];
            if (s > best || (s == best && j < bj)) { best = s; bj = j; }
        }
        sval[threadIdx.x] = best; sidx[threadIdx.x] = bj;
        __syncthreads();
        for (int off = 128; off; off >>= 1) {
            if (threadIdx.x < off) {
                float o = sval[threadIdx.x + off]; int oj = sidx[threadIdx.x + off];
                if (o > sval[threadIdx.x] ||
                    (o == sval[threadIdx.x] && oj >= 0 && oj < sidx[threadIdx.x])) {
                    sval[threadIdx.x] = o; sidx[threadIdx.x] = oj;
                }
            }
            __syncthreads();
        }
        if (threadIdx.x == 0) chosen[r] = sidx[0];
        __syncthreads();
    }
    if (threadIdx.x < KNN) idx[i * KNN + threadIdx.x] = chosen[threadIdx.x];
}

// W += 0.5 at (i,j) and (j,i) for each directed kNN edge (exact in fp32)
__global__ void scatter_w_kernel(const int* __restrict__ idx, float* __restrict__ W)
{
    int id = blockIdx.x * blockDim.x + threadIdx.x;
    if (id >= NPTS * KNN) return;
    int i = id / KNN;
    int j = idx[id];
    atomicAdd(&W[(size_t)i * NPTS + j], 0.5f);
    atomicAdd(&W[(size_t)j * NPTS + i], 0.5f);
}

__global__ __launch_bounds__(256) void degree_dis_kernel(const float* __restrict__ W, float* __restrict__ dis)
{
    __shared__ float red[256];
    const int i = blockIdx.x;
    float s = 0.f;
    for (int j = threadIdx.x; j < NPTS; j += 256) s += W[(size_t)i * NPTS + j];
    red[threadIdx.x] = s; __syncthreads();
    for (int off = 128; off; off >>= 1) {
        if (threadIdx.x < off) red[threadIdx.x] += red[threadIdx.x + off];
        __syncthreads();
    }
    if (threadIdx.x == 0) dis[i] = rsqrtf(fmaxf(red[0], 1e-12f));
}

// M = I + D^-1/2 W D^-1/2  (f16).  Smallest eig of L=I-S <-> largest of M=I+S.
__global__ void build_mh_kernel(const float* __restrict__ W, const float* __restrict__ dis,
                                _Float16* __restrict__ Mh)
{
    size_t id = (size_t)blockIdx.x * blockDim.x + threadIdx.x;
    if (id >= (size_t)NPTS * NPTS) return;
    int i = (int)(id / NPTS), j = (int)(id % NPTS);
    float m = dis[i] * W[id] * dis[j] + (i == j ? 1.0f : 0.0f);
    Mh[id] = (_Float16)m;
}

__global__ void init_v_kernel(float* __restrict__ V)
{
    int id = blockIdx.x * blockDim.x + threadIdx.x;
    if (id >= NPTS * KEIG) return;
    unsigned u = (unsigned)id * 2654435761u;
    u ^= u >> 16; u *= 2246822519u; u ^= u >> 13;
    V[id] = ((float)(u & 0xFFFFFF) / 8388608.0f) - 1.0f;
}

__global__ __launch_bounds__(128) void add_diag_kernel(float* __restrict__ G, float eps)
{
    G[threadIdx.x * KEIG + threadIdx.x] += eps;
}

// in-place upper Cholesky of KEIG x KEIG Gram (single block, 128 threads)
__global__ __launch_bounds__(128) void cholesky_kernel(float* __restrict__ G)
{
    const int t = threadIdx.x;
    for (int k = 0; k < KEIG; ++k) {
        if (t == 0) G[k * KEIG + k] = sqrtf(fmaxf(G[k * KEIG + k], 1e-20f));
        __syncthreads();
        float rkk = G[k * KEIG + k];
        if (t > k) G[k * KEIG + t] /= rkk;
        __syncthreads();
        if (t > k) {
            float rkj = G[k * KEIG + t];
            for (int i = k + 1; i <= t; ++i) G[i * KEIG + t] -= G[k * KEIG + i] * rkj;
        }
        __syncthreads();
    }
}

// V = V2 * R^-1  (R upper-triangular from Cholesky-QR); one thread per row n
__global__ void trisolve_kernel(const float* __restrict__ V2, const float* __restrict__ R,
                                float* __restrict__ V)
{
    int n = blockIdx.x * blockDim.x + threadIdx.x;
    if (n >= NPTS) return;
    const float* src = V2 + (size_t)n * KEIG;
    float*       dst = V  + (size_t)n * KEIG;
    for (int j = 0; j < KEIG; ++j) {
        float v = src[j];
        for (int i = 0; i < j; ++i) v -= dst[i] * R[i * KEIG + j];
        dst[j] = v / R[j * KEIG + j];
    }
}

// lambda_L[k] = 2 - Rayleigh quotient diag (H = V^T M V)
__global__ __launch_bounds__(128) void evals_kernel(const float* __restrict__ H, float* __restrict__ ev)
{
    int t = threadIdx.x;
    ev[t] = 2.0f - H[t * KEIG + t];
}

// resolvent mask, rows indexed by ey (target), gamma = 0.5
__global__ __launch_bounds__(128) void mask_kernel(const float* __restrict__ ex,
                                                   const float* __restrict__ ey,
                                                   float* __restrict__ Dm)
{
    __shared__ float red[128];
    const int t = threadIdx.x;
    red[t] = fmaxf(ex[t], ey[t]); __syncthreads();
    for (int off = 64; off; off >>= 1) {
        if (t < off) red[t] = fmaxf(red[t], red[t + off]);
        __syncthreads();
    }
    const float scale = fmaxf(red[0], 1e-12f);
    const float g1 = sqrtf(fmaxf(ex[t], 0.f) / scale);
    for (int i = 0; i < KEIG; ++i) {
        float g2 = sqrtf(fmaxf(ey[i], 0.f) / scale);
        float re = g2 / (g2 * g2 + 1.f) - g1 / (g1 * g1 + 1.f);
        float im = 1.f / (g2 * g2 + 1.f) - 1.f / (g1 * g1 + 1.f);
        Dm[i * KEIG + t] = re * re + im * im;
    }
}

// per-row regularized solve: (AAt + LMBDA*diag(Dm_row)) x = RHS_row
__global__ __launch_bounds__(128) void ge_solve_kernel(const float* __restrict__ AAt,
                                                       const float* __restrict__ Dm,
                                                       const float* __restrict__ RHS,
                                                       float* __restrict__ Cout,
                                                       float* __restrict__ scratch)
{
    const int row = blockIdx.x;
    const int i   = threadIdx.x;
    float* Mr = scratch + (size_t)row * KEIG * KEIG;
    __shared__ float rhs[KEIG];
    __shared__ float x[KEIG];

    for (int j = 0; j < KEIG; ++j)
        Mr[i * KEIG + j] = AAt[i * KEIG + j] + ((i == j) ? LMBDA * Dm[row * KEIG + i] : 0.f);
    rhs[i] = RHS[row * KEIG + i];
    __syncthreads();

    for (int k = 0; k < KEIG - 1; ++k) {
        const float piv = Mr[k * KEIG + k];
        if (i > k) {
            const float f = Mr[i * KEIG + k] / piv;
            for (int j = k; j < KEIG; ++j) Mr[i * KEIG + j] -= f * Mr[k * KEIG + j];
            rhs[i] -= f * rhs[k];
        }
        __syncthreads();
    }
    for (int k = KEIG - 1; k >= 0; --k) {
        if (i == k) {
            float v = rhs[k];
            for (int j = k + 1; j < KEIG; ++j) v -= Mr[k * KEIG + j] * x[j];
            x[k] = v / Mr[k * KEIG + k];
        }
        __syncthreads();
    }
    Cout[row * KEIG + i] = x[i];
}

// loss pass 1: per-block (row) partial sums of [bij, orth, lap]
__global__ __launch_bounds__(128) void loss_partial_kernel(const float* __restrict__ Cxy,
                                                           const float* __restrict__ Cyx,
                                                           const float* __restrict__ ex,
                                                           const float* __restrict__ ey,
                                                           float* __restrict__ partial)
{
    const int i = blockIdx.x, j = threadIdx.x;
    __shared__ float cxr[KEIG], cyr[KEIG];
    cxr[j] = Cxy[i * KEIG + j];
    cyr[j] = Cyx[i * KEIG + j];
    __syncthreads();
    float p1 = 0, p2 = 0, q1 = 0, q2 = 0;
    for (int k = 0; k < KEIG; ++k) {
        p1 += cxr[k] * Cyx[k * KEIG + j];
        p2 += cyr[k] * Cxy[k * KEIG + j];
        q1 += Cxy[k * KEIG + i] * Cxy[k * KEIG + j];
        q2 += Cyx[k * KEIG + i] * Cyx[k * KEIG + j];
    }
    const float d  = (i == j) ? 1.f : 0.f;
    const float b0 = (p1 - d) * (p1 - d) + (p2 - d) * (p2 - d);
    const float b1 = (q1 - d) * (q1 - d) + (q2 - d) * (q2 - d);
    const float a  = cxr[j] * (ex[j] - ey[i]);
    const float b  = cyr[j] * (ey[j] - ex[i]);
    const float b2 = a * a + b * b;
    __shared__ float r0[KEIG], r1[KEIG], r2[KEIG];
    r0[j] = b0; r1[j] = b1; r2[j] = b2; __syncthreads();
    for (int off = 64; off; off >>= 1) {
        if (j < off) { r0[j] += r0[j + off]; r1[j] += r1[j + off]; r2[j] += r2[j + off]; }
        __syncthreads();
    }
    if (j == 0) { partial[i * 3 + 0] = r0[0]; partial[i * 3 + 1] = r1[0]; partial[i * 3 + 2] = r2[0]; }
}

// loss pass 2: deterministic final reduction + weights
__global__ __launch_bounds__(128) void loss_final_kernel(const float* __restrict__ partial,
                                                         float* __restrict__ out)
{
    __shared__ float r0[KEIG], r1[KEIG], r2[KEIG];
    const int t = threadIdx.x;
    r0[t] = partial[t * 3 + 0];
    r1[t] = partial[t * 3 + 1];
    r2[t] = partial[t * 3 + 2];
    __syncthreads();
    for (int off = 64; off; off >>= 1) {
        if (t < off) { r0[t] += r0[t + off]; r1[t] += r1[t + off]; r2[t] += r2[t + off]; }
        __syncthreads();
    }
    if (t == 0) {
        out[0] = 1.0f   * r0[0];
        out[1] = 1.0f   * r1[0];
        out[2] = 0.001f * r2[0];
    }
}

// ---------------------------------------------------------------------------
// Host orchestration
// ---------------------------------------------------------------------------
extern "C" void kernel_launch(void* const* d_in, const int* in_sizes, int n_in,
                              void* d_out, int out_size, void* d_ws, size_t ws_size,
                              hipStream_t stream)
{
    const float* feat_v = (const float*)d_in[0];
    const float* feat_t = (const float*)d_in[1];
    float* out = (float*)d_out;

    // ---- workspace carve-up (bytes, 256-aligned) ----
    char* base = (char*)d_ws;
    size_t off = 0;
    auto take = [&](size_t bytes) -> char* {
        char* p = base + off;
        off = (off + bytes + 255) & ~(size_t)255;
        return p;
    };
    const size_t NN = (size_t)NPTS * NPTS;
    float*    G    = (float*)   take(NN * 4);                 // Gram / reused as W
    _Float16* Mh   = (_Float16*)take(NN * 2);                 // f16 iteration matrix
    _Float16* FH   = (_Float16*)take((size_t)NPTS * DFEAT * 2);
    float*    sq   = (float*)   take(NPTS * 4);
    float*    dis  = (float*)   take(NPTS * 4);
    int*      idx  = (int*)     take(NPTS * KNN * 4);
    float*    V    = (float*)   take((size_t)NPTS * KEIG * 4);
    float*    V2   = (float*)   take((size_t)NPTS * KEIG * 4);
    _Float16* Vh   = (_Float16*)take((size_t)NPTS * KEIG * 2);
    _Float16* V2h  = (_Float16*)take((size_t)NPTS * KEIG * 2);
    float*    Gs   = (float*)   take(KEIG * KEIG * 4);        // Gram / Cholesky R
    float*    Hm   = (float*)   take(KEIG * KEIG * 4);        // Rayleigh-Ritz
    float*    evV  = (float*)   take(KEIG * 4);
    float*    evT  = (float*)   take(KEIG * 4);
    float*    Af   = (float*)   take(KEIG * DFEAT * 4);
    float*    Bf   = (float*)   take(KEIG * DFEAT * 4);
    _Float16* Ah   = (_Float16*)take(KEIG * DFEAT * 2);
    _Float16* Bh   = (_Float16*)take(KEIG * DFEAT * 2);
    float*    AAt  = (float*)   take(KEIG * KEIG * 4);
    float*    BAt  = (float*)   take(KEIG * KEIG * 4);
    float*    BBt  = (float*)   take(KEIG * KEIG * 4);
    float*    ABt  = (float*)   take(KEIG * KEIG * 4);
    float*    Dxy  = (float*)   take(KEIG * KEIG * 4);
    float*    Dyx  = (float*)   take(KEIG * KEIG * 4);
    float*    Cxy  = (float*)   take(KEIG * KEIG * 4);
    float*    Cyx  = (float*)   take(KEIG * KEIG * 4);
    float*    gesc = (float*)   take((size_t)KEIG * KEIG * KEIG * 4);
    float*    lpar = (float*)   take(KEIG * 3 * 4);
    (void)ws_size; (void)n_in; (void)in_sizes; (void)out_size;

    const dim3 blk128(128), blk256(256);
    auto cdiv = [](int a, int b) { return (a + b - 1) / b; };

    // ---- one spectral phase per feature set ----
    auto spectral = [&](const float* feat, float* evals, float* Aout) {
        cvt_f32_f16_kernel<<<cdiv(NPTS * DFEAT, 256), blk256, 0, stream>>>(feat, FH, NPTS * DFEAT);
        sqnorm_kernel<<<NPTS, blk256, 0, stream>>>(feat, sq);
        // Gram G = FH * FH^T  (3072x3072, K=768) -- WMMA
        wmma_gemm_f16f32<false, true><<<dim3(NPTS / 32, NPTS / 32), blk128, 0, stream>>>(
            FH, FH, G, DFEAT, DFEAT, DFEAT, NPTS);
        // kNN selection, symmetric W = 0.5*(A + A^T) (reuse G buffer)
        knn_select_kernel<<<NPTS, blk256, 0, stream>>>(G, sq, idx);
        hipMemsetAsync(G, 0, NN * 4, stream);
        scatter_w_kernel<<<cdiv(NPTS * KNN, 256), blk256, 0, stream>>>(idx, G);
        // normalized adjacency shifted: M = I + D^-1/2 W D^-1/2 (f16)
        degree_dis_kernel<<<NPTS, blk256, 0, stream>>>(G, dis);
        build_mh_kernel<<<(int)((NN + 255) / 256), blk256, 0, stream>>>(G, dis, Mh);
        // subspace iteration for the KEIG largest eigenpairs of M
        init_v_kernel<<<cdiv(NPTS * KEIG, 256), blk256, 0, stream>>>(V);
        for (int it = 0; it < SUB_ITERS; ++it) {
            cvt_f32_f16_kernel<<<cdiv(NPTS * KEIG, 256), blk256, 0, stream>>>(V, Vh, NPTS * KEIG);
            wmma_gemm_f16f32<false, false><<<dim3(KEIG / 32, NPTS / 32), blk128, 0, stream>>>(
                Mh, Vh, V2, NPTS, NPTS, KEIG, KEIG);            // V2 = M V
            cvt_f32_f16_kernel<<<cdiv(NPTS * KEIG, 256), blk256, 0, stream>>>(V2, V2h, NPTS * KEIG);
            wmma_gemm_f16f32<true, false><<<dim3(KEIG / 32, KEIG / 32), blk128, 0, stream>>>(
                V2h, V2h, Gs, NPTS, KEIG, KEIG, KEIG);          // Gs = V2^T V2
            add_diag_kernel<<<1, blk128, 0, stream>>>(Gs, 1e-6f);
            cholesky_kernel<<<1, blk128, 0, stream>>>(Gs);      // Gs := R (upper)
            trisolve_kernel<<<cdiv(NPTS, 256), blk256, 0, stream>>>(V2, Gs, V); // V = V2 R^-1
        }
        // Rayleigh-Ritz diag: H = V^T (M V);  lambda_L = 2 - diag(H)
        cvt_f32_f16_kernel<<<cdiv(NPTS * KEIG, 256), blk256, 0, stream>>>(V, Vh, NPTS * KEIG);
        wmma_gemm_f16f32<false, false><<<dim3(KEIG / 32, NPTS / 32), blk128, 0, stream>>>(
            Mh, Vh, V2, NPTS, NPTS, KEIG, KEIG);
        cvt_f32_f16_kernel<<<cdiv(NPTS * KEIG, 256), blk256, 0, stream>>>(V2, V2h, NPTS * KEIG);
        wmma_gemm_f16f32<true, false><<<dim3(KEIG / 32, KEIG / 32), blk128, 0, stream>>>(
            Vh, V2h, Hm, NPTS, KEIG, KEIG, KEIG);
        evals_kernel<<<1, blk128, 0, stream>>>(Hm, evals);
        // spectral coefficients: A = V^T F   (128 x 768, K=3072) -- WMMA
        wmma_gemm_f16f32<true, false><<<dim3(DFEAT / 32, KEIG / 32), blk128, 0, stream>>>(
            Vh, FH, Aout, NPTS, KEIG, DFEAT, DFEAT);
    };

    spectral(feat_v, evV, Af);
    spectral(feat_t, evT, Bf);

    // ---- functional map solves ----
    cvt_f32_f16_kernel<<<cdiv(KEIG * DFEAT, 256), blk256, 0, stream>>>(Af, Ah, KEIG * DFEAT);
    cvt_f32_f16_kernel<<<cdiv(KEIG * DFEAT, 256), blk256, 0, stream>>>(Bf, Bh, KEIG * DFEAT);
    wmma_gemm_f16f32<false, true><<<dim3(KEIG / 32, KEIG / 32), blk128, 0, stream>>>(
        Ah, Ah, AAt, DFEAT, DFEAT, DFEAT, KEIG);
    wmma_gemm_f16f32<false, true><<<dim3(KEIG / 32, KEIG / 32), blk128, 0, stream>>>(
        Bh, Ah, BAt, DFEAT, DFEAT, DFEAT, KEIG);
    wmma_gemm_f16f32<false, true><<<dim3(KEIG / 32, KEIG / 32), blk128, 0, stream>>>(
        Bh, Bh, BBt, DFEAT, DFEAT, DFEAT, KEIG);
    wmma_gemm_f16f32<false, true><<<dim3(KEIG / 32, KEIG / 32), blk128, 0, stream>>>(
        Ah, Bh, ABt, DFEAT, DFEAT, DFEAT, KEIG);

    mask_kernel<<<1, blk128, 0, stream>>>(evV, evT, Dxy);  // rows ~ target (t)
    mask_kernel<<<1, blk128, 0, stream>>>(evT, evV, Dyx);

    ge_solve_kernel<<<KEIG, blk128, 0, stream>>>(AAt, Dxy, BAt, Cxy, gesc);
    ge_solve_kernel<<<KEIG, blk128, 0, stream>>>(BBt, Dyx, ABt, Cyx, gesc);

    // ---- loss (deterministic two-stage reduction) ----
    loss_partial_kernel<<<KEIG, blk128, 0, stream>>>(Cxy, Cyx, evV, evT, lpar);
    loss_final_kernel<<<1, blk128, 0, stream>>>(lpar, out);
}
